// InfoNCEGraph_59261958750666
// MI455X (gfx1250) — matrix operand order repaired and checked by reading, not compiled, not run on hardware
//
#include <hip/hip_runtime.h>
#include <hip/hip_bf16.h>

// ---------------------------------------------------------------------------
// InfoNCEGraph pipeline for MI455X (gfx1250, wave32, WMMA bf16, async->LDS)
// N=2048, C=256, MEM=32768, CLS=60, P=128, NN=512, T=0.8
// ---------------------------------------------------------------------------

#define N_BATCH 2048
#define IN_C    256
#define MEM_N   32768
#define P_NUM   128
#define NN_NUM  512
#define T_INV   1.25f   // 1/0.8

typedef __attribute__((ext_vector_type(16))) __bf16 v16bf;
typedef __attribute__((ext_vector_type(8)))  float  v8f;

__device__ __forceinline__ unsigned short f2bf(float x) {
    unsigned int u = __float_as_uint(x);
    unsigned int r = u + 0x7FFFu + ((u >> 16) & 1u);   // round-to-nearest-even
    return (unsigned short)(r >> 16);
}

// A fragment: 16x32 bf16, row = m0 + (lane&15).
// Layout: lanes 0-15 hold K=k0+0..15 contiguous, lanes 16-31 hold K=k0+16..31 (matches B)
__device__ __forceinline__ v16bf load_frag_a(const unsigned short* mat, int ld,
                                             int row0, int k0, int lane) {
    int m = row0 + (lane & 15);
    int h = lane >> 4;
    const unsigned short* r = mat + (size_t)m * ld;
    union { v16bf v; uint4 q[2]; } f;
    f.q[0] = *reinterpret_cast<const uint4*>(r + k0 + 16 * h);
    f.q[1] = *reinterpret_cast<const uint4*>(r + k0 + 16 * h + 8);
    return f.v;
}

// B fragment: 32x16 bf16, column n = col0 + (lane&15) is row n of the stored matrix.
// Layout (matches A): lanes 0-15 hold K=k0+0..15 contiguous, lanes 16-31 hold K=k0+16..31
__device__ __forceinline__ v16bf load_frag_b(const unsigned short* mat, int ld,
                                             int col0, int k0, int lane) {
    int n = col0 + (lane & 15);
    int h = lane >> 4;
    const unsigned short* r = mat + (size_t)n * ld;
    const uint4* p = reinterpret_cast<const uint4*>(r + k0 + 16 * h);
    union { v16bf v; uint4 q[2]; } f;
    f.q[0] = p[0];
    f.q[1] = p[1];
    return f.v;
}

// B fragment from an LDS-staged 16-row x 512B tile (row-contiguous).
__device__ __forceinline__ v16bf load_frag_b_lds(const uint4* stagePtr, int k0, int lane) {
    int n = lane & 15;
    int h = lane >> 4;
    // byte addr = n*512 + k0*2 + h*32  ->  uint4 index = n*32 + k0/8 + 2h
    const uint4* p = stagePtr + n * 32 + (k0 >> 3) + 2 * h;
    union { v16bf v; uint4 q[2]; } f;
    f.q[0] = p[0];
    f.q[1] = p[1];
    return f.v;
}

// Async copy of one 8 KB bank tile (16 rows x 512 B, contiguous) into this
// wave's LDS stage buffer. INST_OFFSET is added to BOTH the LDS and global
// addresses (ISA 08_async_tensor 4.4), so a single lane*16 offset VGPR plus
// offset:i*512 walks the whole tile. 16 x (32 lanes x b128) = 8192 B.
__device__ __forceinline__ void async_tile_load(const unsigned short* gbase,
                                                unsigned vlds, unsigned voff) {
    unsigned long long sbase = (unsigned long long)(size_t)gbase;
    asm volatile(
        "global_load_async_to_lds_b128 %0, %1, %2 offset:0\n"
        "global_load_async_to_lds_b128 %0, %1, %2 offset:512\n"
        "global_load_async_to_lds_b128 %0, %1, %2 offset:1024\n"
        "global_load_async_to_lds_b128 %0, %1, %2 offset:1536\n"
        "global_load_async_to_lds_b128 %0, %1, %2 offset:2048\n"
        "global_load_async_to_lds_b128 %0, %1, %2 offset:2560\n"
        "global_load_async_to_lds_b128 %0, %1, %2 offset:3072\n"
        "global_load_async_to_lds_b128 %0, %1, %2 offset:3584\n"
        "global_load_async_to_lds_b128 %0, %1, %2 offset:4096\n"
        "global_load_async_to_lds_b128 %0, %1, %2 offset:4608\n"
        "global_load_async_to_lds_b128 %0, %1, %2 offset:5120\n"
        "global_load_async_to_lds_b128 %0, %1, %2 offset:5632\n"
        "global_load_async_to_lds_b128 %0, %1, %2 offset:6144\n"
        "global_load_async_to_lds_b128 %0, %1, %2 offset:6656\n"
        "global_load_async_to_lds_b128 %0, %1, %2 offset:7168\n"
        "global_load_async_to_lds_b128 %0, %1, %2 offset:7680\n"
        :: "v"(vlds), "v"(voff), "s"(sbase) : "memory");
}

__device__ __forceinline__ void wait_asynccnt0() {
    asm volatile("s_wait_asynccnt 0x0" ::: "memory");
}

#define WMMA_BF16(a, b, c) \
    __builtin_amdgcn_wmma_f32_16x16x32_bf16(false, (a), false, (b), (short)0, (c), false, false)

// ---------------------------------------------------------------------------
// K0: float -> bf16 conversion (grid-strided)
// ---------------------------------------------------------------------------
__global__ void cvt_f32_bf16_k(const float* __restrict__ s,
                               unsigned short* __restrict__ d, int n) {
    int i  = blockIdx.x * blockDim.x + threadIdx.x;
    int st = gridDim.x * blockDim.x;
    for (; i < n; i += st) d[i] = f2bf(s[i]);
}

__global__ void copy_f32_k(const float* __restrict__ s, float* __restrict__ d, int n) {
    int i  = blockIdx.x * blockDim.x + threadIdx.x;
    int st = gridDim.x * blockDim.x;
    for (; i < n; i += st) d[i] = s[i];
}

// ---------------------------------------------------------------------------
// K1: Y = normalize(X @ W^T + b) in bf16.  16 rows per workgroup, 8 waves.
// ---------------------------------------------------------------------------
__global__ void __launch_bounds__(256)
proj_norm_k(const unsigned short* __restrict__ X, const unsigned short* __restrict__ Wm,
            const float* __restrict__ bias, unsigned short* __restrict__ out) {
    __shared__ float ytile[16][256];
    __shared__ float sumsq[16];
    int tid = threadIdx.x, lane = tid & 31, wave = tid >> 5;
    int m0 = blockIdx.x * 16;

    if (tid < 16) sumsq[tid] = 0.0f;
    __syncthreads();

    v16bf a[8];
#pragma unroll
    for (int kk = 0; kk < 8; kk++) a[kk] = load_frag_a(X, 256, m0, kk * 32, lane);

    for (int jt = wave; jt < 16; jt += 8) {
        int c0 = jt * 16;
        v8f c = {};
#pragma unroll
        for (int kk = 0; kk < 8; kk++) {
            v16bf b = load_frag_b(Wm, 256, c0, kk * 32, lane);
            c = WMMA_BF16(a[kk], b, c);
        }
        int col = c0 + (lane & 15);
        int h = lane >> 4;
        float bb = bias[col];
#pragma unroll
        for (int v = 0; v < 8; v++) {
            int r = v + 8 * h;
            float y = c[v] + bb;
            ytile[r][col] = y;
            atomicAdd(&sumsq[r], y * y);
        }
    }
    __syncthreads();

    for (int i = tid; i < 16 * 256; i += 256) {
        int r = i >> 8, col = i & 255;
        float rn = rsqrtf(sumsq[r] + 1e-24f);
        out[(size_t)(m0 + r) * 256 + col] = f2bf(ytile[r][col] * rn);
    }
}

// ---------------------------------------------------------------------------
// K2: nn_idx[n] = argmax_m  fn[n] . q[m]   (deterministic: smallest idx wins ties)
// ---------------------------------------------------------------------------
__global__ void __launch_bounds__(256)
nn_argmax_k(const unsigned short* __restrict__ FN, const unsigned short* __restrict__ Q,
            int* __restrict__ nn_idx) {
    __shared__ unsigned long long best[16];
    int tid = threadIdx.x, lane = tid & 31, wave = tid >> 5;
    int m0 = blockIdx.x * 16;
    if (tid < 16) best[tid] = 0ull;
    __syncthreads();

    v16bf a[8];
#pragma unroll
    for (int kk = 0; kk < 8; kk++) a[kk] = load_frag_a(FN, 256, m0, kk * 32, lane);

    for (int jt = wave; jt < N_BATCH / 16; jt += 8) {
        int c0 = jt * 16;
        v8f c = {};
#pragma unroll
        for (int kk = 0; kk < 8; kk++) {
            v16bf b = load_frag_b(Q, 256, c0, kk * 32, lane);
            c = WMMA_BF16(a[kk], b, c);
        }
        int h = lane >> 4;
        int mm = c0 + (lane & 15);
#pragma unroll
        for (int v = 0; v < 8; v++) {
            float s = c[v];
            unsigned int bbits = __float_as_uint(s);
            bbits ^= (bbits & 0x80000000u) ? 0xFFFFFFFFu : 0x80000000u;  // order-preserving
            unsigned long long key =
                ((unsigned long long)bbits << 32) |
                (unsigned long long)(0xFFFFFFFFu - (unsigned)mm);        // smaller idx wins ties
            atomicMax(&best[v + 8 * h], key);
        }
    }
    __syncthreads();
    if (tid < 16)
        nn_idx[m0 + tid] = (int)(0xFFFFFFFFu - (unsigned)(best[tid] & 0xFFFFFFFFull));
}

// ---------------------------------------------------------------------------
// K3: sg[n] = q[nn_idx[n]]; scatter sg into bank2 (last-writer-wins, deterministic)
// ---------------------------------------------------------------------------
__global__ void sg_scatter_k(const unsigned short* __restrict__ Q,
                             const int* __restrict__ nn_idx,
                             const int* __restrict__ input_index,
                             unsigned short* __restrict__ SG,
                             unsigned short* __restrict__ BANK2,
                             float* __restrict__ flag2, int n) {
    int i = blockIdx.x * blockDim.x + threadIdx.x;
    if (i >= n) return;
    int src = nn_idx[i];
    const uint4* q4 = reinterpret_cast<const uint4*>(Q + (size_t)src * 256);
    uint4* s4 = reinterpret_cast<uint4*>(SG + (size_t)i * 256);
    for (int k = 0; k < 32; k++) s4[k] = q4[k];

    int ii = input_index[i];
    bool last = true;
    for (int j = i + 1; j < n; j++)
        if (input_index[j] == ii) { last = false; break; }
    if (last) {
        uint4* b4 = reinterpret_cast<uint4*>(BANK2 + (size_t)ii * 256);
        for (int k = 0; k < 32; k++) b4[k] = q4[k];
        flag2[ii] = 1.0f;
    }
}

// ---------------------------------------------------------------------------
// K4: fused all_pairs GEMM + top-k selection + InfoNCE per-row loss.
// One workgroup per 16 rows; two WMMA passes over bank2 (L2-resident, 16 MB).
// Each wave async-copies its 16x256 bank tile (8 KB) into a private LDS stage
// buffer (global_load_async_to_lds_b128 / s_wait_asynccnt), feeds WMMA from
// ds_load_b128.  Pass 1 builds histograms (512-bin pos sims / neg sims, 256-bin
// rand_u among negs); a threshold scan picks radix cut points; pass 2 gathers
// the P smallest positives and accumulates logsumexp sums (anchored at sim=1,
// valid since |sim|<=1) for hard and random negatives.
// LDS: 64K stage + 64K sim hists + 16K u hist + 8K pos + <1K scalars ~ 153 KB.
// ---------------------------------------------------------------------------
__global__ void __launch_bounds__(256)
infonce_main_k(const unsigned short* __restrict__ SG,
               const unsigned short* __restrict__ BANK2,
               const float* __restrict__ flag2,
               const int* __restrict__ label,
               const int* __restrict__ label_all,
               const float* __restrict__ rand_u,
               float* __restrict__ row_loss, float* __restrict__ row_valid) {
    __shared__ uint4        stage[8][512];          // 8 waves x 8 KB tile
    __shared__ unsigned int hist_pos[16][512];
    __shared__ unsigned int hist_neg[16][512];
    __shared__ unsigned int hist_u[16][256];
    __shared__ float        pos_vals[16][128];
    __shared__ unsigned int pos_n[16], capPc[16], capNc[16], capUc[16];
    __shared__ int          bp[16], bn[16], bu[16];
    __shared__ unsigned int capP[16], capN[16], capU[16];
    __shared__ float        S_h[16], S_r[16];
    __shared__ unsigned int posTot[16], negTot[16];
    __shared__ int          rlabel[16];

    int tid = threadIdx.x, lane = tid & 31, wave = tid >> 5;
    int m0 = blockIdx.x * 16;

    for (int i = tid; i < 16 * 512; i += 256) {
        (&hist_pos[0][0])[i] = 0u;
        (&hist_neg[0][0])[i] = 0u;
    }
    for (int i = tid; i < 16 * 256; i += 256) (&hist_u[0][0])[i] = 0u;
    if (tid < 16) {
        pos_n[tid] = 0u; capPc[tid] = 0u; capNc[tid] = 0u; capUc[tid] = 0u;
        S_h[tid] = 0.0f; S_r[tid] = 0.0f;
        rlabel[tid] = label[m0 + tid];
    }
    __syncthreads();

    v16bf a[8];
#pragma unroll
    for (int kk = 0; kk < 8; kk++) a[kk] = load_frag_a(SG, 256, m0, kk * 32, lane);

    const uint4* myStage = &stage[wave][0];
    // LDS flat address low 32 bits == LDS offset on amdgcn
    unsigned vlds = (unsigned)(size_t)myStage + (unsigned)lane * 16u;
    unsigned voff = (unsigned)lane * 16u;

    // ---------------- pass 1: histograms ----------------
    for (int mt = wave; mt < MEM_N / 16; mt += 8) {
        int c0 = mt * 16;
        async_tile_load(BANK2 + (size_t)c0 * 256, vlds, voff);
        if (c0 + 128 + 16 <= MEM_N)  // L2 prefetch of the wave's next bank tile
            __builtin_prefetch(BANK2 + (size_t)(c0 + 128 + (lane & 15)) * 256, 0, 1);
        wait_asynccnt0();
        v8f c = {};
#pragma unroll
        for (int kk = 0; kk < 8; kk++) {
            v16bf b = load_frag_b_lds(myStage, kk * 32, lane);
            c = WMMA_BF16(a[kk], b, c);
        }
        int col = c0 + (lane & 15);
        int h = lane >> 4;
        int la = label_all[col];
        bool vb = flag2[col] > 0.0f;
#pragma unroll
        for (int v = 0; v < 8; v++) {
            int r = v + 8 * h;
            float s = c[v];
            int bin = (int)((s + 1.0f) * 256.0f);
            bin = bin < 0 ? 0 : (bin > 511 ? 511 : bin);
            if (vb) {
                if (rlabel[r] == la) {
                    atomicAdd(&hist_pos[r][bin], 1u);
                } else {
                    atomicAdd(&hist_neg[r][bin], 1u);
                    float u = rand_u[(size_t)(m0 + r) * MEM_N + col];
                    int ub = (int)(u * 256.0f);
                    ub = ub < 0 ? 0 : (ub > 255 ? 255 : ub);
                    atomicAdd(&hist_u[r][ub], 1u);
                }
            }
        }
    }
    __syncthreads();

    // ---------------- threshold scan (1 thread / row) ----------------
    if (tid < 16) {
        int r = tid;
        unsigned tot = 0;
        for (int b2 = 0; b2 < 512; b2++) tot += hist_pos[r][b2];
        posTot[r] = tot;
        unsigned cum = 0; int bb = 511; unsigned cap = 0xFFFFFFFFu;
        for (int b2 = 0; b2 < 512; b2++) {            // ascending: P smallest positives
            unsigned hc = hist_pos[r][b2];
            if (cum + hc >= (unsigned)P_NUM) { bb = b2; cap = (unsigned)P_NUM - cum; break; }
            cum += hc;
        }
        bp[r] = bb; capP[r] = cap;

        tot = 0;
        for (int b2 = 0; b2 < 512; b2++) tot += hist_neg[r][b2];
        negTot[r] = tot;
        cum = 0; bb = 0; cap = 0xFFFFFFFFu;
        for (int b2 = 511; b2 >= 0; b2--) {           // descending: NN largest negatives
            unsigned hc = hist_neg[r][b2];
            if (cum + hc >= (unsigned)NN_NUM) { bb = b2; cap = (unsigned)NN_NUM - cum; break; }
            cum += hc;
        }
        bn[r] = bb; capN[r] = cap;

        cum = 0; bb = 0; cap = 0xFFFFFFFFu;
        for (int b2 = 255; b2 >= 0; b2--) {           // descending: top-NN of rand_u among negs
            unsigned hc = hist_u[r][b2];
            if (cum + hc >= (unsigned)NN_NUM) { bb = b2; cap = (unsigned)NN_NUM - cum; break; }
            cum += hc;
        }
        bu[r] = bb; capU[r] = cap;
    }
    __syncthreads();

    // ---------------- pass 2: gather + accumulate ----------------
    for (int mt = wave; mt < MEM_N / 16; mt += 8) {
        int c0 = mt * 16;
        async_tile_load(BANK2 + (size_t)c0 * 256, vlds, voff);
        wait_asynccnt0();
        v8f c = {};
#pragma unroll
        for (int kk = 0; kk < 8; kk++) {
            v16bf b = load_frag_b_lds(myStage, kk * 32, lane);
            c = WMMA_BF16(a[kk], b, c);
        }
        int col = c0 + (lane & 15);
        int h = lane >> 4;
        int la = label_all[col];
        bool vb = flag2[col] > 0.0f;
#pragma unroll
        for (int v = 0; v < 8; v++) {
            int r = v + 8 * h;
            float s = c[v];
            int bin = (int)((s + 1.0f) * 256.0f);
            bin = bin < 0 ? 0 : (bin > 511 ? 511 : bin);
            if (vb) {
                if (rlabel[r] == la) {
                    bool take = bin < bp[r];
                    if (!take && bin == bp[r]) take = atomicAdd(&capPc[r], 1u) < capP[r];
                    if (take) {
                        unsigned i2 = atomicAdd(&pos_n[r], 1u);
                        if (i2 < (unsigned)P_NUM) pos_vals[r][i2] = s;
                    }
                } else {
                    float e = __expf((s - 1.0f) * T_INV);  // anchored logsumexp term
                    bool th = bin > bn[r];
                    if (!th && bin == bn[r]) th = atomicAdd(&capNc[r], 1u) < capN[r];
                    if (th) atomicAdd(&S_h[r], e);
                    float u = rand_u[(size_t)(m0 + r) * MEM_N + col];
                    int ub = (int)(u * 256.0f);
                    ub = ub < 0 ? 0 : (ub > 255 ? 255 : ub);
                    bool tr = ub > bu[r];
                    if (!tr && ub == bu[r]) tr = atomicAdd(&capUc[r], 1u) < capU[r];
                    if (tr) atomicAdd(&S_r[r], e);
                }
            }
        }
    }
    __syncthreads();

    // ---------------- per-row loss ----------------
    for (int r = wave; r < 16; r += 8) {
        unsigned cnt = pos_n[r];
        if (cnt > (unsigned)P_NUM) cnt = (unsigned)P_NUM;
        float lseh = T_INV + __logf(S_h[r] + 1e-30f);  // LSE(negs/T), max anchored at sim=1
        float lser = T_INV + __logf(S_r[r] + 1e-30f);
        float acc = 0.0f;
        for (unsigned i = lane; i < cnt; i += 32) {
            float pt = pos_vals[r][i] * T_INV;
            acc += log1pf(__expf(lseh - pt)) + log1pf(__expf(lser - pt));
        }
#pragma unroll
        for (int off = 16; off > 0; off >>= 1) acc += __shfl_down(acc, off, 32);
        if (lane == 0) {
            row_loss[m0 + r]  = acc;
            row_valid[m0 + r] = (posTot[r] >= (unsigned)P_NUM &&
                                 negTot[r] >= (unsigned)(2 * NN_NUM)) ? 1.0f : 0.0f;
        }
    }
}

// ---------------------------------------------------------------------------
// K5: final reduction -> scalar
// ---------------------------------------------------------------------------
__global__ void __launch_bounds__(256)
reduce_k(const float* __restrict__ row_loss, const float* __restrict__ row_valid,
         float* __restrict__ out) {
    __shared__ float ssum[256];
    __shared__ float scnt[256];
    int tid = threadIdx.x;
    float s = 0.0f, c = 0.0f;
    for (int i = tid; i < N_BATCH; i += 256) {
        float v = row_valid[i];
        s += v * row_loss[i];
        c += v;
    }
    ssum[tid] = s; scnt[tid] = c;
    __syncthreads();
    for (int off = 128; off > 0; off >>= 1) {
        if (tid < off) { ssum[tid] += ssum[tid + off]; scnt[tid] += scnt[tid + off]; }
        __syncthreads();
    }
    if (tid == 0) {
        float cnt = scnt[0];
        float denom = (cnt < 1.0f ? 1.0f : cnt) * (2.0f * (float)P_NUM);
        out[0] = (cnt > 0.0f) ? ssum[0] / denom : 0.0f;
    }
}

// ---------------------------------------------------------------------------
// Launch
// ---------------------------------------------------------------------------
extern "C" void kernel_launch(void* const* d_in, const int* in_sizes, int n_in,
                              void* d_out, int out_size, void* d_ws, size_t ws_size,
                              hipStream_t stream) {
    const float* f          = (const float*)d_in[0];
    const float* g          = (const float*)d_in[1];
    const float* Wm         = (const float*)d_in[2];
    const float* bias       = (const float*)d_in[3];
    const float* bank       = (const float*)d_in[4];
    const float* bank_flag  = (const float*)d_in[5];
    const float* rand_u     = (const float*)d_in[6];
    const int*   label_all  = (const int*)d_in[7];
    const int*   label      = (const int*)d_in[8];
    const int*   input_index = (const int*)d_in[9];
    float* out = (float*)d_out;

    char* ws = (char*)d_ws;
    size_t off = 0;
    auto alloc = [&](size_t bytes) {
        size_t o = off;
        off = (off + bytes + 255) & ~(size_t)255;
        return o;
    };
    const size_t SZ_XBF = (size_t)N_BATCH * IN_C * 2;
    size_t oFbf   = alloc(SZ_XBF);                         // f bf16
    size_t oGbf   = alloc(SZ_XBF);                         // g bf16
    size_t oWbf   = alloc((size_t)IN_C * IN_C * 2);        // W bf16
    size_t oFNbf  = alloc(SZ_XBF);                         // f_n bf16
    size_t oQbf   = alloc(SZ_XBF);                         // q bf16
    size_t oSGbf  = alloc(SZ_XBF);                         // sg bf16
    size_t oBKbf  = alloc((size_t)MEM_N * IN_C * 2);       // bank2 bf16
    size_t oFlag2 = alloc((size_t)MEM_N * 4);              // flag2
    size_t oNN    = alloc((size_t)N_BATCH * 4);            // nn_idx
    size_t oLoss  = alloc((size_t)N_BATCH * 4);            // per-row loss
    size_t oValid = alloc((size_t)N_BATCH * 4);            // per-row valid
    if (off > ws_size) return;                             // workspace too small

    unsigned short* Fbf  = (unsigned short*)(ws + oFbf);
    unsigned short* Gbf  = (unsigned short*)(ws + oGbf);
    unsigned short* Wbf  = (unsigned short*)(ws + oWbf);
    unsigned short* FNbf = (unsigned short*)(ws + oFNbf);
    unsigned short* Qbf  = (unsigned short*)(ws + oQbf);
    unsigned short* SGbf = (unsigned short*)(ws + oSGbf);
    unsigned short* BKbf = (unsigned short*)(ws + oBKbf);
    float* flag2 = (float*)(ws + oFlag2);
    int*   nnIdx = (int*)(ws + oNN);
    float* rloss = (float*)(ws + oLoss);
    float* rvalid = (float*)(ws + oValid);

    const int XE = N_BATCH * IN_C;           // 524288
    const int WE = IN_C * IN_C;              // 65536
    const int BE = MEM_N * IN_C;             // 8388608

    cvt_f32_bf16_k<<<1024, 256, 0, stream>>>(f, Fbf, XE);
    cvt_f32_bf16_k<<<1024, 256, 0, stream>>>(g, Gbf, XE);
    cvt_f32_bf16_k<<<256, 256, 0, stream>>>(Wm, Wbf, WE);
    cvt_f32_bf16_k<<<4096, 256, 0, stream>>>(bank, BKbf, BE);
    copy_f32_k<<<128, 256, 0, stream>>>(bank_flag, flag2, MEM_N);

    proj_norm_k<<<N_BATCH / 16, 256, 0, stream>>>(Fbf, Wbf, bias, FNbf);
    proj_norm_k<<<N_BATCH / 16, 256, 0, stream>>>(Gbf, Wbf, bias, Qbf);

    nn_argmax_k<<<N_BATCH / 16, 256, 0, stream>>>(FNbf, Qbf, nnIdx);

    sg_scatter_k<<<N_BATCH / 256, 256, 0, stream>>>(Qbf, nnIdx, input_index,
                                                    SGbf, BKbf, flag2, N_BATCH);

    infonce_main_k<<<N_BATCH / 16, 256, 0, stream>>>(SGbf, BKbf, flag2, label,
                                                     label_all, rand_u, rloss, rvalid);

    reduce_k<<<1, 256, 0, stream>>>(rloss, rvalid, out);
}
